// Net_7378753815409
// MI455X (gfx1250) — matrix operand order repaired and checked by reading, not compiled, not run on hardware
//
#include <hip/hip_runtime.h>

#define NS 512      // sites
#define NB 512      // batch
#define NB2 1024    // augmented batch (2B)
#define HH 128      // hidden
#define LN_EPS 1e-5f

typedef __attribute__((ext_vector_type(16))) _Float16 v16h;
typedef __attribute__((ext_vector_type(8)))  _Float16 v8h;
typedef __attribute__((ext_vector_type(8)))  float    v8f;
typedef __attribute__((ext_vector_type(4)))  unsigned int uint32x4;
typedef __attribute__((ext_vector_type(8)))  int      int32x8;
typedef __attribute__((ext_vector_type(4)))  int      int32x4;

// Tensor Data Mover availability (6-arg builtin form on this toolchain).
#if defined(__HIP_DEVICE_COMPILE__) && defined(__gfx1250__) &&                 \
    __has_builtin(__builtin_amdgcn_tensor_load_to_lds) &&                      \
    __has_builtin(__builtin_amdgcn_tensor_store_from_lds) &&                   \
    __has_builtin(__builtin_amdgcn_s_wait_tensorcnt)
#define USE_TDM 1
#else
#define USE_TDM 0
#endif

#if USE_TDM
// LDS byte offset of a shared-memory object (generic -> addrspace(3) cast).
static __device__ __forceinline__ unsigned lds_off(const void* p) {
  return (unsigned)(uintptr_t)(const __attribute__((address_space(3))) void*)p;
}

// Build a 2-D TDM descriptor (D#) per CDNA5 ISA ch.8: tile_h lines of tile_w
// f16 elements; consecutive lines stride0 elements apart in global memory;
// LDS destination/source is contiguous.
static __device__ __forceinline__ void tdm_desc(unsigned lds_byte_addr,
                                                const void* gptr,
                                                unsigned tile_w, unsigned tile_h,
                                                unsigned stride0,
                                                uint32x4* g0, int32x8* g1) {
  unsigned long long ga = (unsigned long long)(uintptr_t)gptr;
  uint32x4 a;
  a[0] = 1u;                                   // count=1, user mode, no gather
  a[1] = lds_byte_addr;                        // bits 63:32  lds_addr
  a[2] = (unsigned)(ga & 0xffffffffu);         // bits 95:64  global_addr lo
  a[3] = (unsigned)(ga >> 32) | (2u << 30);    // addr hi, type=2 ("image")
  int32x8 b;
  b[0] = (int)(1u << 16);                      // data_size=1 -> 2 bytes, mask=0
  b[1] = (int)((tile_w & 0xffffu) << 16);      // tensor_dim0[15:0] @63:48
  b[2] = (int)(((tile_w >> 16) & 0xffffu) |    // tensor_dim0[31:16] @79:64
               ((tile_h & 0xffffu) << 16));    // tensor_dim1[15:0] @95:80
  b[3] = (int)(((tile_h >> 16) & 0xffffu) |    // tensor_dim1[31:16]
               ((tile_w & 0xffffu) << 16));    // tile_dim0 @127:112
  b[4] = (int)(tile_h & 0xffffu);              // tile_dim1 @143:128, tile_dim2=0
  b[5] = (int)stride0;                         // tensor_dim0_stride[31:0]
  b[6] = 0;                                    // stride0 hi, dim1_stride lo
  b[7] = 0;
  *g0 = a;
  *g1 = b;
}
#endif

static __device__ __forceinline__ v8f wmma_f16(v16h a, v16h b, v8f c) {
  // D = A(16x32 f16) x B(32x16 f16) + C(16x16 f32)
  return __builtin_amdgcn_wmma_f32_16x16x32_f16(false, a, false, b, (short)0, c,
                                                false, false);
}

// A-fragment (16x32 f16) from row-major [.., ldk] storage at (row0, k0)=base.
// Lanes rows M=lane&15 hold K=kb..kb+15 contiguous,
// kb = (lane>>4)*16.
template <typename P>
static __device__ __forceinline__ v16h load_a16(P base, int ldk, int lane) {
  const int mr = lane & 15;
  const int kb = (lane >> 4) << 4;
  auto p = base + mr * ldk + kb;
  v8h lo = *(const v8h*)(p);
  v8h hi = *(const v8h*)(p + 8);
  v16h a;
#pragma unroll
  for (int i = 0; i < 8; ++i) { a[i] = lo[i]; a[i + 8] = hi[i]; }
  return a;
}

// B-fragment (32x16 f16) for y = x @ W^T, W row-major [128][128] f16.
// B[k][n] = W[n][k]; lane column n = lane&15, K = ko..ko+15 contiguous,
// ko = kc*32 + (lane>>4)*16.
static __device__ __forceinline__ v16h load_b16(const _Float16* __restrict__ w,
                                                int ntile, int kc, int lane) {
  const int nr = (ntile << 4) + (lane & 15);
  const int ko = (kc << 5) + ((lane >> 4) << 4);
  return *(const v16h*)(w + nr * HH + ko);
}

// ---------------------------------------------------------------------------
// prep: convert six 128x128 weights to f16, zero hidden site-0 slice and the
// per-batch accumulators.
// ---------------------------------------------------------------------------
__global__ __launch_bounds__(256) void prep_kernel(
    const float* __restrict__ wih, const float* __restrict__ wuh,
    const float* __restrict__ wp1, const float* __restrict__ wp2,
    const float* __restrict__ wa1, const float* __restrict__ wa2,
    _Float16* __restrict__ w16, _Float16* __restrict__ hid,
    float* __restrict__ acc_log, float* __restrict__ acc_phase) {
  int idx = blockIdx.x * 256 + threadIdx.x;
  if (idx < HH * HH) {
    w16[0 * HH * HH + idx] = (_Float16)wih[idx];
    w16[1 * HH * HH + idx] = (_Float16)wuh[idx];
    w16[2 * HH * HH + idx] = (_Float16)wp1[idx];
    w16[3 * HH * HH + idx] = (_Float16)wp2[idx];
    w16[4 * HH * HH + idx] = (_Float16)wa1[idx];
    w16[5 * HH * HH + idx] = (_Float16)wa2[idx];
  }
  if (idx < NB2 * HH) {
    int row = idx >> 7, col = idx & 127;
    hid[(size_t)row * NS * HH + col] = (_Float16)0.f;  // rolled site 0 = 0
  }
  if (idx < NB2) { acc_log[idx] = 0.f; acc_phase[idx] = 0.f; }
}

// ---------------------------------------------------------------------------
// GRU scan: one wave32 block per 16 augmented-batch rows (64 blocks total).
// h (16x128 f16) lives in LDS; each step does 2 GEMMs h@W^T via 64 WMMAs,
// gates + LayerNorm in-register (shfl_xor row reductions). The rolled store
// hidden[row][t+1][:] goes out through the Tensor Data Mover so it overlaps
// the next step's WMMAs; TENSORcnt is drained one step later (WAR on hsm).
// ---------------------------------------------------------------------------
__global__ __launch_bounds__(32) void gru_scan_kernel(
    const float* __restrict__ inp,
    const float* __restrict__ ii_w, const float* __restrict__ ii_b,
    const float* __restrict__ ih_b,
    const float* __restrict__ ui_w, const float* __restrict__ ui_b,
    const float* __restrict__ uh_b,
    const float* __restrict__ ln_g, const float* __restrict__ ln_b,
    const _Float16* __restrict__ wih, const _Float16* __restrict__ wuh,
    _Float16* __restrict__ hid) {
  __shared__ _Float16 hsm[16 * HH];
  __shared__ float xw[32];

  const int lane = threadIdx.x;
  const int half = lane >> 4;
  const int l16  = lane & 15;
  const int R0   = blockIdx.x * 16;

  // hoist per-column params for this lane's 8 output columns (one per n-tile)
  float c_iiB[8], c_ihB[8], c_uiB[8], c_uhB[8];
  float c_wi0[8], c_wi1[8], c_wu0[8], c_wu1[8], c_g[8], c_bb[8];
#pragma unroll
  for (int n = 0; n < 8; ++n) {
    int nc = (n << 4) + l16;
    c_iiB[n] = ii_b[nc];     c_ihB[n] = ih_b[nc];
    c_uiB[n] = ui_b[nc];     c_uhB[n] = uh_b[nc];
    c_wi0[n] = ii_w[2 * nc]; c_wi1[n] = ii_w[2 * nc + 1];
    c_wu0[n] = ui_w[2 * nc]; c_wu1[n] = ui_w[2 * nc + 1];
    c_g[n] = ln_g[nc];       c_bb[n] = ln_b[nc];
  }

  v8f acc_i[8], acc_u[8];

  for (int t = 0; t < NS - 1; ++t) {
    // stage x_t for this block's 16 rows (symmetry-augmented indexing)
    if (lane < 16) {
      int b = R0 + lane, sx = t;
      if (b >= NB) { b -= NB; sx = NS - 1 - t; }
      xw[lane]      = inp[(b * 2 + 0) * NS + sx];
      xw[16 + lane] = inp[(b * 2 + 1) * NS + sx];
    }
    asm volatile("s_wait_dscnt 0x0" ::: "memory");

    float x0r[8], x1r[8];
#pragma unroll
    for (int i = 0; i < 8; ++i) {
      int mr = (half << 3) + i;
      x0r[i] = xw[mr];
      x1r[i] = xw[16 + mr];
    }

#pragma unroll
    for (int n = 0; n < 8; ++n)
#pragma unroll
      for (int i = 0; i < 8; ++i) { acc_i[n][i] = 0.f; acc_u[n][i] = 0.f; }

    if (t > 0) {  // recurrent terms h @ ih_w^T, h @ uh_w^T
#pragma unroll
      for (int kc = 0; kc < 4; ++kc) {
        v16h a = load_a16((const _Float16*)&hsm[kc * 32], HH, lane);
#pragma unroll
        for (int n = 0; n < 8; ++n) {
          acc_i[n] = wmma_f16(a, load_b16(wih, n, kc, lane), acc_i[n]);
          acc_u[n] = wmma_f16(a, load_b16(wuh, n, kc, lane), acc_u[n]);
        }
      }
    }

    // gates + candidate (in-place into acc_i), accumulate row sums for LN
    const float rb = (t > 0) ? 1.f : 0.f;
    float rsum[8];
#pragma unroll
    for (int i = 0; i < 8; ++i) rsum[i] = 0.f;
#pragma unroll
    for (int n = 0; n < 8; ++n) {
      int nc = (n << 4) + l16;
      float bi = c_iiB[n] + rb * c_ihB[n];
      float bu = c_uiB[n] + rb * c_uhB[n];
#pragma unroll
      for (int i = 0; i < 8; ++i) {
        int mr = (half << 3) + i;
        float pig = acc_i[n][i] + bi + x0r[i] * c_wi0[n] + x1r[i] * c_wi1[n];
        float pug = acc_u[n][i] + bu + x0r[i] * c_wu0[n] + x1r[i] * c_wu1[n];
        float ig = tanhf(pig);
        float ug = 1.f / (1.f + __expf(-pug));
        float hold = (t > 0) ? (float)hsm[mr * HH + nc] : 0.f;
        float v = hold * (1.f - ug) + ig * ug;
        acc_i[n][i] = v;
        rsum[i] += v;
      }
    }

    // LayerNorm across H=128: row = (half,i); butterfly over 16 lanes
    float mu[8], rstd[8], vs[8];
#pragma unroll
    for (int i = 0; i < 8; ++i) {
#pragma unroll
      for (int m = 8; m >= 1; m >>= 1) rsum[i] += __shfl_xor(rsum[i], m, 16);
      mu[i] = rsum[i] * (1.f / 128.f);
      vs[i] = 0.f;
    }
#pragma unroll
    for (int n = 0; n < 8; ++n)
#pragma unroll
      for (int i = 0; i < 8; ++i) {
        float d = acc_i[n][i] - mu[i];
        vs[i] += d * d;
      }
#pragma unroll
    for (int i = 0; i < 8; ++i) {
#pragma unroll
      for (int m = 8; m >= 1; m >>= 1) vs[i] += __shfl_xor(vs[i], m, 16);
      rstd[i] = rsqrtf(vs[i] * (1.f / 128.f) + LN_EPS);
    }

#if USE_TDM
    // drain last step's TDM store before overwriting the LDS h tile (WAR)
    __builtin_amdgcn_s_wait_tensorcnt(0);
#endif

    // write normalized h back to LDS (f16) for the next step's A-fragments
#pragma unroll
    for (int n = 0; n < 8; ++n) {
      int nc = (n << 4) + l16;
#pragma unroll
      for (int i = 0; i < 8; ++i) {
        int mr = (half << 3) + i;
        float hn = (acc_i[n][i] - mu[i]) * rstd[i] * c_g[n] + c_bb[n];
        hsm[mr * HH + nc] = (_Float16)hn;
      }
    }
    asm volatile("s_wait_dscnt 0x0" ::: "memory");

    // rolled store: h(step t) -> hidden[row][t+1][:]
#if USE_TDM
    {
      uint32x4 g0; int32x8 g1;
      int32x4 z4 = (int32x4)0;
      int32x8 z8 = (int32x8)0;
      tdm_desc(lds_off(hsm),
               hid + (((size_t)R0 * NS) + (t + 1)) * HH,
               HH, 16, (unsigned)(NS * HH), &g0, &g1);
      __builtin_amdgcn_tensor_store_from_lds(g0, g1, z4, z4, z8, 0);
    }
#else
    {
      int r = lane >> 1;
      int off = (lane & 1) << 6;  // 0 or 64 halves
      const uint4* src = (const uint4*)&hsm[r * HH + off];
      uint4* dst = (uint4*)(hid + (((size_t)(R0 + r) * NS) + (t + 1)) * HH + off);
#pragma unroll
      for (int q = 0; q < 8; ++q) dst[q] = src[q];
    }
#endif
  }
}

// ---------------------------------------------------------------------------
// Fused readout: one wave32 block per 16 (b,s) rows. TDM-stage the 16x128 f16
// A tile into LDS, then four H x H GEMMs (128 WMMAs), 2-wide heads,
// softmax/log/atan2 and per-batch site reduction via float atomics.
// ---------------------------------------------------------------------------
__global__ __launch_bounds__(32) void conv_head_kernel(
    const float* __restrict__ inp, const _Float16* __restrict__ hid,
    const _Float16* __restrict__ w1, const float* __restrict__ b1,
    const _Float16* __restrict__ w2, const float* __restrict__ b2,
    const _Float16* __restrict__ w3, const float* __restrict__ b3,
    const _Float16* __restrict__ w4, const float* __restrict__ b4,
    const float* __restrict__ prw, const float* __restrict__ prb,
    const float* __restrict__ snw, const float* __restrict__ snb,
    const float* __restrict__ csw, const float* __restrict__ csb,
    float* __restrict__ acc_log, float* __restrict__ acc_phase) {
  __shared__ _Float16 ain[16 * HH];
  __shared__ _Float16 mid[16 * HH];
  __shared__ float rowdat[16][6];

  const int lane = threadIdx.x;
  const int half = lane >> 4;
  const int l16  = lane & 15;
  const size_t row0 = (size_t)blockIdx.x * 16;  // row = b*NS + s
  const _Float16* Ain = hid + row0 * HH;

  // stage A tile (16 x 128 f16, contiguous 4KB) into LDS
#if USE_TDM
  {
    uint32x4 g0; int32x8 g1;
    int32x4 z4 = (int32x4)0;
    int32x8 z8 = (int32x8)0;
    tdm_desc(lds_off(ain), Ain, HH, 16, HH, &g0, &g1);
    __builtin_amdgcn_tensor_load_to_lds(g0, g1, z4, z4, z8, 0);
    __builtin_amdgcn_s_wait_tensorcnt(0);
  }
#else
  {
    int r = lane >> 1;
    int off = (lane & 1) << 6;
    const uint4* src = (const uint4*)(Ain + r * HH + off);
    uint4* dst = (uint4*)&ain[r * HH + off];
#pragma unroll
    for (int q = 0; q < 8; ++q) dst[q] = src[q];
  }
  asm volatile("s_wait_dscnt 0x0" ::: "memory");
#endif

  v8f acc[8];

  // ---- chain A, layer 1: mid = relu(ain @ w1^T + b1)
#pragma unroll
  for (int n = 0; n < 8; ++n)
#pragma unroll
    for (int i = 0; i < 8; ++i) acc[n][i] = 0.f;
#pragma unroll
  for (int kc = 0; kc < 4; ++kc) {
    v16h a = load_a16((const _Float16*)&ain[kc * 32], HH, lane);
#pragma unroll
    for (int n = 0; n < 8; ++n) acc[n] = wmma_f16(a, load_b16(w1, n, kc, lane), acc[n]);
  }
#pragma unroll
  for (int n = 0; n < 8; ++n) {
    int nc = (n << 4) + l16;
    float bb = b1[nc];
#pragma unroll
    for (int i = 0; i < 8; ++i)
      mid[((half << 3) + i) * HH + nc] = (_Float16)fmaxf(acc[n][i] + bb, 0.f);
  }
  asm volatile("s_wait_dscnt 0x0" ::: "memory");

  // ---- chain A, layer 2 + pr head
#pragma unroll
  for (int n = 0; n < 8; ++n)
#pragma unroll
    for (int i = 0; i < 8; ++i) acc[n][i] = 0.f;
#pragma unroll
  for (int kc = 0; kc < 4; ++kc) {
    v16h a = load_a16((const _Float16*)&mid[kc * 32], HH, lane);
#pragma unroll
    for (int n = 0; n < 8; ++n) acc[n] = wmma_f16(a, load_b16(w2, n, kc, lane), acc[n]);
  }
#pragma unroll
  for (int i = 0; i < 8; ++i) {
    float s0 = 0.f, s1 = 0.f;
#pragma unroll
    for (int n = 0; n < 8; ++n) {
      int nc = (n << 4) + l16;
      float v = fmaxf(acc[n][i] + b2[nc], 0.f);
      s0 += v * prw[nc];
      s1 += v * prw[HH + nc];
    }
#pragma unroll
    for (int m = 8; m >= 1; m >>= 1) {
      s0 += __shfl_xor(s0, m, 16);
      s1 += __shfl_xor(s1, m, 16);
    }
    if (l16 == 0) {
      rowdat[(half << 3) + i][0] = s0 + prb[0];
      rowdat[(half << 3) + i][1] = s1 + prb[1];
    }
  }

  // ---- chain B, layer 1: mid = relu(ain @ w3^T + b3)
#pragma unroll
  for (int n = 0; n < 8; ++n)
#pragma unroll
    for (int i = 0; i < 8; ++i) acc[n][i] = 0.f;
#pragma unroll
  for (int kc = 0; kc < 4; ++kc) {
    v16h a = load_a16((const _Float16*)&ain[kc * 32], HH, lane);
#pragma unroll
    for (int n = 0; n < 8; ++n) acc[n] = wmma_f16(a, load_b16(w3, n, kc, lane), acc[n]);
  }
#pragma unroll
  for (int n = 0; n < 8; ++n) {
    int nc = (n << 4) + l16;
    float bb = b3[nc];
#pragma unroll
    for (int i = 0; i < 8; ++i)
      mid[((half << 3) + i) * HH + nc] = (_Float16)fmaxf(acc[n][i] + bb, 0.f);
  }
  asm volatile("s_wait_dscnt 0x0" ::: "memory");

  // ---- chain B, layer 2 + sin/cos heads
#pragma unroll
  for (int n = 0; n < 8; ++n)
#pragma unroll
    for (int i = 0; i < 8; ++i) acc[n][i] = 0.f;
#pragma unroll
  for (int kc = 0; kc < 4; ++kc) {
    v16h a = load_a16((const _Float16*)&mid[kc * 32], HH, lane);
#pragma unroll
    for (int n = 0; n < 8; ++n) acc[n] = wmma_f16(a, load_b16(w4, n, kc, lane), acc[n]);
  }
#pragma unroll
  for (int i = 0; i < 8; ++i) {
    float ss0 = 0.f, ss1 = 0.f, cc0 = 0.f, cc1 = 0.f;
#pragma unroll
    for (int n = 0; n < 8; ++n) {
      int nc = (n << 4) + l16;
      float v = fmaxf(acc[n][i] + b4[nc], 0.f);
      ss0 += v * snw[nc];
      ss1 += v * snw[HH + nc];
      cc0 += v * csw[nc];
      cc1 += v * csw[HH + nc];
    }
#pragma unroll
    for (int m = 8; m >= 1; m >>= 1) {
      ss0 += __shfl_xor(ss0, m, 16);
      ss1 += __shfl_xor(ss1, m, 16);
      cc0 += __shfl_xor(cc0, m, 16);
      cc1 += __shfl_xor(cc1, m, 16);
    }
    if (l16 == 0) {
      rowdat[(half << 3) + i][2] = ss0 + snb[0];
      rowdat[(half << 3) + i][3] = ss1 + snb[1];
      rowdat[(half << 3) + i][4] = cc0 + csb[0];
      rowdat[(half << 3) + i][5] = cc1 + csb[1];
    }
  }
  asm volatile("s_wait_dscnt 0x0" ::: "memory");

  // ---- finalize rows: softmax over 2 channels, log/phase, atomic reduce
  if (lane < 16) {
    int r = lane;
    size_t grow = row0 + r;
    int b = (int)(grow >> 9);   // / NS
    int s = (int)(grow & 511);  // % NS
    int bs = b, sx = s;
    if (bs >= NB) { bs -= NB; sx = NS - 1 - s; }
    float x0 = inp[(bs * 2 + 0) * NS + sx];
    float x1 = inp[(bs * 2 + 1) * NS + sx];
    float d0 = rowdat[r][0], d1 = rowdat[r][1];
    float sn0 = rowdat[r][2], sn1 = rowdat[r][3];
    float cs0 = rowdat[r][4], cs1 = rowdat[r][5];
    float mx = fmaxf(d0, d1);
    float e0 = __expf(d0 - mx), e1 = __expf(d1 - mx);
    float inv = 1.f / (e0 + e1);
    float pw = logf(e0 * inv * x0 + e1 * inv * x1);
    float ph = atan2f(sn0, cs0) * x0 + atan2f(sn1, cs1) * x1;
    atomicAdd(acc_log + b, pw);
    atomicAdd(acc_phase + b, ph);
  }
}

// ---------------------------------------------------------------------------
// Combine symmetry pair into (log_wf[512], phase_wf[512]) -> d_out (1024 f32).
// ---------------------------------------------------------------------------
__global__ __launch_bounds__(256) void final_kernel(
    const float* __restrict__ acc_log, const float* __restrict__ acc_phase,
    float* __restrict__ out) {
  int b = blockIdx.x * 256 + threadIdx.x;
  if (b < NB) {
    float l0 = acc_log[b], l1 = acc_log[b + NB];
    float p0 = acc_phase[b], p1 = acc_phase[b + NB];
    float mx = fmaxf(l0, l1);
    float w0 = expf(l0 - mx), w1 = expf(l1 - mx);
    out[b] = 0.5f * logf(0.5f * (w0 + w1)) + 0.5f * mx;
    out[NB + b] = atan2f(w0 * sinf(p0) + w1 * sinf(p1),
                         w0 * cosf(p0) + w1 * cosf(p1));
  }
}

extern "C" void kernel_launch(void* const* d_in, const int* in_sizes, int n_in,
                              void* d_out, int out_size, void* d_ws, size_t ws_size,
                              hipStream_t stream) {
  const float* inp   = (const float*)d_in[0];
  const float* ii_w  = (const float*)d_in[1];
  const float* ii_b  = (const float*)d_in[2];
  const float* ih_w  = (const float*)d_in[3];
  const float* ih_b  = (const float*)d_in[4];
  const float* ui_w  = (const float*)d_in[5];
  const float* ui_b  = (const float*)d_in[6];
  const float* uh_w  = (const float*)d_in[7];
  const float* uh_b  = (const float*)d_in[8];
  const float* ln_g  = (const float*)d_in[9];
  const float* ln_b  = (const float*)d_in[10];
  const float* ph1_w = (const float*)d_in[11];
  const float* ph1_b = (const float*)d_in[12];
  const float* ph2_w = (const float*)d_in[13];
  const float* ph2_b = (const float*)d_in[14];
  const float* ah1_w = (const float*)d_in[15];
  const float* ah1_b = (const float*)d_in[16];
  const float* ah2_w = (const float*)d_in[17];
  const float* ah2_b = (const float*)d_in[18];
  const float* pr_w  = (const float*)d_in[19];
  const float* pr_b  = (const float*)d_in[20];
  const float* sin_w = (const float*)d_in[21];
  const float* sin_b = (const float*)d_in[22];
  const float* cos_w = (const float*)d_in[23];
  const float* cos_b = (const float*)d_in[24];

  char* ws = (char*)d_ws;
  _Float16* hid = (_Float16*)ws;  // [1024][512][128] f16 (rolled hidden)
  size_t hid_bytes = (size_t)NB2 * NS * HH * sizeof(_Float16);
  _Float16* w16 = (_Float16*)(ws + hid_bytes);  // 6 x 128x128 f16 weights
  size_t w16_bytes = 6ull * HH * HH * sizeof(_Float16);
  float* acc_log = (float*)(ws + hid_bytes + w16_bytes);
  float* acc_phase = acc_log + NB2;

  prep_kernel<<<512, 256, 0, stream>>>(ih_w, uh_w, ph1_w, ph2_w, ah1_w, ah2_w,
                                       w16, hid, acc_log, acc_phase);
  gru_scan_kernel<<<64, 32, 0, stream>>>(inp, ii_w, ii_b, ih_b, ui_w, ui_b, uh_b,
                                         ln_g, ln_b, w16, w16 + HH * HH, hid);
  conv_head_kernel<<<(NB2 * NS) / 16, 32, 0, stream>>>(
      inp, hid, w16 + 2 * HH * HH, ph1_b, w16 + 3 * HH * HH, ph2_b,
      w16 + 4 * HH * HH, ah1_b, w16 + 5 * HH * HH, ah2_b,
      pr_w, pr_b, sin_w, sin_b, cos_w, cos_b, acc_log, acc_phase);
  final_kernel<<<2, 256, 0, stream>>>(acc_log, acc_phase, (float*)d_out);

  (void)in_sizes; (void)n_in; (void)out_size; (void)ws_size;
}
